// SpecDecoder_86577950753092
// MI455X (gfx1250) — matrix operand order repaired and checked by reading, not compiled, not run on hardware
//
#include <hip/hip_runtime.h>
#include <hip/hip_bf16.h>

typedef __attribute__((ext_vector_type(16))) __bf16 v16bf;
typedef __attribute__((ext_vector_type(8)))  float  v8f;

#define B_TOT    1024
#define H_DIM    512
#define E_DIM    256
#define L_SEQ    128
#define G_DIM    2048      // 4*H
#define K_TOT    768       // E + H
#define BT       32        // batch rows per workgroup (two WMMA M-tiles)
#define NTHREADS 512       // 16 wave32
#define NWAVES   16

__device__ __forceinline__ unsigned short f2bf(float f) {
  unsigned u = __float_as_uint(f);
  u += 0x7FFFu + ((u >> 16) & 1u);          // round-to-nearest-even
  return (unsigned short)(u >> 16);
}
__device__ __forceinline__ float sigf(float x) {
  return 1.0f / (1.0f + __expf(-x));
}
// defeat loop-invariant hoisting of small weight fragments across the t-loop
__device__ __forceinline__ const unsigned short* opaque(const unsigned short* p) {
  asm volatile("" : "+v"(p));
  return p;
}

union Frag { v16bf v; uint4 u[2]; };

// A operand (16x32 bf16): lane holds row m=lane&15; elements j: K=half*16+j
// -> two 16B chunks at +0 and +8 (ushort elements)
__device__ __forceinline__ v16bf load_fragA(const unsigned short* p) {
  Frag f;
  f.u[0] = *(const uint4*)(p);
  f.u[1] = *(const uint4*)(p + 8);
  return f.v;
}
// B operand (32x16 bf16) from row-major weights: chunks at +0 and +8
__device__ __forceinline__ v16bf load_fragB(const unsigned short* p) {
  Frag f;
  f.u[0] = *(const uint4*)(p);
  f.u[1] = *(const uint4*)(p + 8);
  return f.v;
}

// ---- async copy: one 8KB (kk,wave) slice of WgT -> LDS, 16 x b128 per wave.
// INST_OFFSET is added to BOTH global and LDS addresses; WgT is laid out so the
// offsets coincide, so the whole slice needs only immediates 0..7680.
#define ASYNC_CP(ldsdst, voff, src, OFF)                                      \
  asm volatile("global_load_async_to_lds_b128 %0, %1, %2 offset:" #OFF        \
               :: "v"(ldsdst), "v"(voff), "s"(src) : "memory")

__device__ __forceinline__ void issue_async_kk(unsigned ldsdst, unsigned voff,
                                               const unsigned short* src) {
  // async LDS writes are UNORDERED vs DS reads: drain pending ds reads of the
  // buffer being recycled before the DMA may overwrite it.
  asm volatile("s_wait_dscnt 0x0" ::: "memory");
  ASYNC_CP(ldsdst, voff, src, 0);    ASYNC_CP(ldsdst, voff, src, 512);
  ASYNC_CP(ldsdst, voff, src, 1024); ASYNC_CP(ldsdst, voff, src, 1536);
  ASYNC_CP(ldsdst, voff, src, 2048); ASYNC_CP(ldsdst, voff, src, 2560);
  ASYNC_CP(ldsdst, voff, src, 3072); ASYNC_CP(ldsdst, voff, src, 3584);
  ASYNC_CP(ldsdst, voff, src, 4096); ASYNC_CP(ldsdst, voff, src, 4608);
  ASYNC_CP(ldsdst, voff, src, 5120); ASYNC_CP(ldsdst, voff, src, 5632);
  ASYNC_CP(ldsdst, voff, src, 6144); ASYNC_CP(ldsdst, voff, src, 6656);
  ASYNC_CP(ldsdst, voff, src, 7168); ASYNC_CP(ldsdst, voff, src, 7680);
}
#define WAIT_ASYNC16() asm volatile("s_wait_asynccnt 0x10" ::: "memory")
#define WAIT_ASYNC0()  asm volatile("s_wait_asynccnt 0x0" ::: "memory")

// ---------------- prologue: weight conversion + wave-tiled re-layout --------
// WgT element order: [kk 0..23][wave 0..15][instr q' 0..15][lane 0..31][e 0..7]
//   tile q = q'>>1, half-chunk s = q'&1
//   n = (q>>1)*512 + w*32 + (q&1)*16 + (l&15)
//   k = kk*32 + (l>>4)*16 + s*8 + e     (k<256 -> W_ih, else W_hh)
__global__ void SpecDecoder_prep(const float* __restrict__ W_ih,
                                 const float* __restrict__ W_hh,
                                 const float* __restrict__ b_ih,
                                 const float* __restrict__ b_hh,
                                 const float* __restrict__ W1,
                                 const float* __restrict__ W2,
                                 const float* __restrict__ W3,
                                 unsigned short* __restrict__ WgT,
                                 unsigned short* __restrict__ W1b,
                                 unsigned short* __restrict__ W2b,
                                 unsigned short* __restrict__ W3b,
                                 float* __restrict__ bg) {
  const int stride = gridDim.x * blockDim.x;
  const int i0 = blockIdx.x * blockDim.x + threadIdx.x;
  for (int o = i0; o < G_DIM * K_TOT; o += stride) {
    int e  = o & 7;
    int l  = (o >> 3) & 31;
    int s  = (o >> 8) & 1;
    int q  = (o >> 9) & 7;
    int w  = (o >> 12) & 15;
    int kk = o >> 16;
    int n = (q >> 1) * 512 + w * 32 + (q & 1) * 16 + (l & 15);
    int k = kk * 32 + (l >> 4) * 16 + s * 8 + e;
    float v = (k < E_DIM) ? W_ih[n * E_DIM + k] : W_hh[n * H_DIM + (k - E_DIM)];
    WgT[o] = f2bf(v);
  }
  for (int idx = i0; idx < 128 * H_DIM; idx += stride) W1b[idx] = f2bf(W1[idx]);
  for (int idx = i0; idx < 128 * 128;   idx += stride) W2b[idx] = f2bf(W2[idx]);
  for (int idx = i0; idx < E_DIM * 128; idx += stride) W3b[idx] = f2bf(W3[idx]);
  for (int idx = i0; idx < G_DIM;       idx += stride) bg[idx] = b_ih[idx] + b_hh[idx];
}

// ---------------- persistent recurrent kernel ----------------
extern __shared__ __attribute__((aligned(16))) char dynls[];  // 256KB B staging

__global__ __launch_bounds__(NTHREADS, 1)
void SpecDecoder_main(const float* __restrict__ h0,
                      const float* __restrict__ xseq,        // (1024,128,256)
                      const unsigned short* __restrict__ WgT,// wave-tiled gates W
                      const float* __restrict__ bg,          // (2048)
                      const unsigned short* __restrict__ W1b,// (128,512)
                      const float* __restrict__ b1,
                      const unsigned short* __restrict__ W2b,// (128,128)
                      const float* __restrict__ b2,
                      const unsigned short* __restrict__ W3b,// (256,128)
                      const float* __restrict__ b3,
                      float* __restrict__ out) {             // (1024,128,256)
  __shared__ __attribute__((aligned(16))) unsigned short a16[BT * K_TOT]; // [32][768]
  __shared__ __attribute__((aligned(16))) unsigned short z1s[BT * 128];
  __shared__ __attribute__((aligned(16))) unsigned short z2s[BT * 128];

  const int tid  = threadIdx.x;
  const int wv   = tid >> 5;
  const int lane = tid & 31;
  const int half = lane >> 4;
  const int mrow = lane & 15;
  const int ncol = lane & 15;
  const int b0   = blockIdx.x * BT;

  // per-wave private double buffers in dynamic LDS (8KB each)
  unsigned short* const bufA = (unsigned short*)(dynls + wv * 16384);
  unsigned short* const bufB = (unsigned short*)(dynls + wv * 16384 + 8192);
  const unsigned ldsA = (unsigned)(uintptr_t)bufA + lane * 16;
  const unsigned ldsB = (unsigned)(uintptr_t)bufB + lane * 16;
  const unsigned voff = lane * 16;
  const unsigned short* const srcW = WgT + (size_t)wv * 4096;  // + kk*65536 elems

  // ---- init: h0 tile -> bf16 into a16[:,256:768]
  {
    int m  = tid >> 4;
    int c0 = (tid & 15) * 32;
    const float4* src = (const float4*)(h0 + (size_t)(b0 + m) * H_DIM + c0);
    alignas(16) unsigned short tmp[32];
#pragma unroll
    for (int q = 0; q < 8; ++q) {
      float4 v = src[q];
      tmp[q * 4 + 0] = f2bf(v.x); tmp[q * 4 + 1] = f2bf(v.y);
      tmp[q * 4 + 2] = f2bf(v.z); tmp[q * 4 + 3] = f2bf(v.w);
    }
    uint4* dst = (uint4*)&a16[m * K_TOT + E_DIM + c0];
#pragma unroll
    for (int q = 0; q < 4; ++q) dst[q] = ((const uint4*)tmp)[q];
  }
  // ---- start token: out[:,0,:] = one-hot(0)
  for (int idx = tid; idx < BT * E_DIM; idx += NTHREADS) {
    int m = idx >> 8, n = idx & 255;
    out[(size_t)(b0 + m) * (L_SEQ * E_DIM) + n] = (n == 0) ? 1.0f : 0.0f;
  }

  float creg[2][2][8];
#pragma unroll
  for (int mt = 0; mt < 2; ++mt)
#pragma unroll
    for (int p = 0; p < 2; ++p)
#pragma unroll
      for (int r = 0; r < 8; ++r) creg[mt][p][r] = 0.0f;

  float biasg[8];
#pragma unroll
  for (int g = 0; g < 4; ++g)
#pragma unroll
    for (int p = 0; p < 2; ++p)
      biasg[g * 2 + p] = bg[g * H_DIM + wv * 32 + p * 16 + ncol];
  const float b1v = b1[(wv & 7) * 16 + ncol];
  const float b2v = b2[(wv & 7) * 16 + ncol];
  const float b3v = b3[wv * 16 + ncol];

  const unsigned short* const aBase = a16 + mrow * K_TOT + half * 16;
  const int mtW = wv >> 3;
  const int ntW = wv & 7;

  __syncthreads();

  for (int t = 0; t < L_SEQ - 1; ++t) {
    // kick off kk=0 B-slice DMA before x staging to overlap with it
    issue_async_kk(ldsA, voff, srcW);

    // ---- phase 1: stage x_t as bf16 into a16[:,0:256]
    {
      int m  = tid >> 4;
      int c0 = (tid & 15) * 16;
      const float4* src =
          (const float4*)(xseq + ((size_t)(b0 + m) * L_SEQ + t) * E_DIM + c0);
      alignas(16) unsigned short tmp[16];
#pragma unroll
      for (int q = 0; q < 4; ++q) {
        float4 v = src[q];
        tmp[q * 4 + 0] = f2bf(v.x); tmp[q * 4 + 1] = f2bf(v.y);
        tmp[q * 4 + 2] = f2bf(v.z); tmp[q * 4 + 3] = f2bf(v.w);
      }
      uint4* dst = (uint4*)&a16[m * K_TOT + c0];
      dst[0] = ((const uint4*)tmp)[0];
      dst[1] = ((const uint4*)tmp)[1];
    }
    __syncthreads();

    // ---- phase 2: gates = [x|h] @ Wg^T + bg   (M=32, N=2048, K=768)
    v8f acc[2][8];
#pragma unroll
    for (int mt = 0; mt < 2; ++mt)
#pragma unroll
      for (int i = 0; i < 8; ++i)
#pragma unroll
        for (int r = 0; r < 8; ++r) acc[mt][i][r] = biasg[i];

    for (int kk2 = 0; kk2 < 12; ++kk2) {
      const int kk = kk2 * 2;
      // issue kk+1 into bufB, then consume kk from bufA
      issue_async_kk(ldsB, voff, srcW + (size_t)(kk + 1) * 65536);
      WAIT_ASYNC16();
      {
        const int ko = kk * 32;
        v16bf A0 = load_fragA(aBase + ko);
        v16bf A1 = load_fragA(aBase + 16 * K_TOT + ko);
#pragma unroll
        for (int q = 0; q < 8; ++q) {
          Frag Bf;
          Bf.u[0] = *(const uint4*)(bufA + q * 512 + lane * 8);
          Bf.u[1] = *(const uint4*)(bufA + q * 512 + 256 + lane * 8);
          acc[0][q] = __builtin_amdgcn_wmma_f32_16x16x32_bf16(
              false, A0, false, Bf.v, (short)0, acc[0][q], false, false);
          acc[1][q] = __builtin_amdgcn_wmma_f32_16x16x32_bf16(
              false, A1, false, Bf.v, (short)0, acc[1][q], false, false);
        }
      }
      // issue kk+2 into bufA (except last pair), then consume kk+1 from bufB
      if (kk2 < 11) {
        issue_async_kk(ldsA, voff, srcW + (size_t)(kk + 2) * 65536);
        WAIT_ASYNC16();
      } else {
        WAIT_ASYNC0();
      }
      {
        const int ko = (kk + 1) * 32;
        v16bf A0 = load_fragA(aBase + ko);
        v16bf A1 = load_fragA(aBase + 16 * K_TOT + ko);
#pragma unroll
        for (int q = 0; q < 8; ++q) {
          Frag Bf;
          Bf.u[0] = *(const uint4*)(bufB + q * 512 + lane * 8);
          Bf.u[1] = *(const uint4*)(bufB + q * 512 + 256 + lane * 8);
          acc[0][q] = __builtin_amdgcn_wmma_f32_16x16x32_bf16(
              false, A0, false, Bf.v, (short)0, acc[0][q], false, false);
          acc[1][q] = __builtin_amdgcn_wmma_f32_16x16x32_bf16(
              false, A1, false, Bf.v, (short)0, acc[1][q], false, false);
        }
      }
    }

    // ---- phase 3: LSTM cell update entirely in accumulator registers
#pragma unroll
    for (int mt = 0; mt < 2; ++mt) {
#pragma unroll
      for (int p = 0; p < 2; ++p) {
#pragma unroll
        for (int r = 0; r < 8; ++r) {
          float iv = acc[mt][0 + p][r];
          float fv = acc[mt][2 + p][r];
          float gv = acc[mt][4 + p][r];
          float ov = acc[mt][6 + p][r];
          float c  = sigf(fv) * creg[mt][p][r] + sigf(iv) * tanhf(gv);
          creg[mt][p][r] = c;
          float h  = sigf(ov) * tanhf(c);
          int m = mt * 16 + r + 8 * half;
          int j = wv * 32 + p * 16 + ncol;
          a16[m * K_TOT + E_DIM + j] = f2bf(h);
        }
      }
    }
    __syncthreads();

    // ---- phase 4: z1 = relu(h @ W1^T + b1)  (M=32, N=128, K=512), 16 waves
    {
      v8f az;
#pragma unroll
      for (int r = 0; r < 8; ++r) az[r] = b1v;
      const unsigned short* ap = a16 + (mtW * 16 + mrow) * K_TOT + E_DIM + half * 16;
      const unsigned short* bp =
          opaque(W1b + (size_t)(ntW * 16 + ncol) * H_DIM + half * 16);
#pragma unroll 4
      for (int kk = 0; kk < H_DIM / 32; ++kk) {
        v16bf Av = load_fragA(ap);
        v16bf Bv = load_fragB(bp);
        az = __builtin_amdgcn_wmma_f32_16x16x32_bf16(
            false, Av, false, Bv, (short)0, az, false, false);
        ap += 32; bp += 32;
      }
#pragma unroll
      for (int r = 0; r < 8; ++r) {
        float zv = az[r] > 0.0f ? az[r] : 0.0f;
        z1s[(mtW * 16 + r + 8 * half) * 128 + ntW * 16 + ncol] = f2bf(zv);
      }
    }
    __syncthreads();

    // ---- phase 5: z2 = relu(z1 @ W2^T + b2)  (M=32, N=128, K=128), 16 waves
    {
      v8f az;
#pragma unroll
      for (int r = 0; r < 8; ++r) az[r] = b2v;
      const unsigned short* ap = z1s + (mtW * 16 + mrow) * 128 + half * 16;
      const unsigned short* bp =
          opaque(W2b + (size_t)(ntW * 16 + ncol) * 128 + half * 16);
#pragma unroll
      for (int kk = 0; kk < 4; ++kk) {
        v16bf Av = load_fragA(ap);
        v16bf Bv = load_fragB(bp);
        az = __builtin_amdgcn_wmma_f32_16x16x32_bf16(
            false, Av, false, Bv, (short)0, az, false, false);
        ap += 32; bp += 32;
      }
#pragma unroll
      for (int r = 0; r < 8; ++r) {
        float zv = az[r] > 0.0f ? az[r] : 0.0f;
        z2s[(mtW * 16 + r + 8 * half) * 128 + ntW * 16 + ncol] = f2bf(zv);
      }
    }
    __syncthreads();

    // ---- phase 6: tok = z2 @ W3^T + b3  (M=32, N=256, K=128)
    {
      v8f az0, az1;
#pragma unroll
      for (int r = 0; r < 8; ++r) { az0[r] = b3v; az1[r] = b3v; }
      const unsigned short* ap0 = z2s + mrow * 128 + half * 16;
      const unsigned short* ap1 = z2s + (16 + mrow) * 128 + half * 16;
      const unsigned short* bp =
          opaque(W3b + (size_t)(wv * 16 + ncol) * 128 + half * 16);
#pragma unroll
      for (int kk = 0; kk < 4; ++kk) {
        v16bf Bv = load_fragB(bp);
        v16bf A0 = load_fragA(ap0);
        v16bf A1 = load_fragA(ap1);
        az0 = __builtin_amdgcn_wmma_f32_16x16x32_bf16(
            false, A0, false, Bv, (short)0, az0, false, false);
        az1 = __builtin_amdgcn_wmma_f32_16x16x32_bf16(
            false, A1, false, Bv, (short)0, az1, false, false);
        ap0 += 32; ap1 += 32; bp += 32;
      }
#pragma unroll
      for (int r = 0; r < 8; ++r) {
        int m0 = r + 8 * half;
        out[((size_t)(b0 + m0) * L_SEQ + (t + 1)) * E_DIM + wv * 16 + ncol] = az0[r];
        out[((size_t)(b0 + 16 + m0) * L_SEQ + (t + 1)) * E_DIM + wv * 16 + ncol] = az1[r];
      }
    }
    __syncthreads();
  }
}

extern "C" void kernel_launch(void* const* d_in, const int* in_sizes, int n_in,
                              void* d_out, int out_size, void* d_ws, size_t ws_size,
                              hipStream_t stream) {
  (void)in_sizes; (void)n_in; (void)out_size; (void)ws_size;
  const float* h0   = (const float*)d_in[0];
  const float* xseq = (const float*)d_in[1];
  const float* W_ih = (const float*)d_in[2];
  const float* W_hh = (const float*)d_in[3];
  const float* b_ih = (const float*)d_in[4];
  const float* b_hh = (const float*)d_in[5];
  const float* W1   = (const float*)d_in[6];
  const float* b1   = (const float*)d_in[7];
  const float* W2   = (const float*)d_in[8];
  const float* b2   = (const float*)d_in[9];
  const float* W3   = (const float*)d_in[10];
  const float* b3   = (const float*)d_in[11];

  char* ws = (char*)d_ws;
  unsigned short* WgT = (unsigned short*)ws; ws += (size_t)G_DIM * K_TOT * 2;
  unsigned short* W1b = (unsigned short*)ws; ws += (size_t)128 * H_DIM * 2;
  unsigned short* W2b = (unsigned short*)ws; ws += (size_t)128 * 128 * 2;
  unsigned short* W3b = (unsigned short*)ws; ws += (size_t)E_DIM * 128 * 2;
  float* bg = (float*)ws;                    ws += (size_t)G_DIM * 4;

  SpecDecoder_prep<<<512, 256, 0, stream>>>(W_ih, W_hh, b_ih, b_hh, W1, W2, W3,
                                            WgT, W1b, W2b, W3b, bg);
  SpecDecoder_main<<<B_TOT / BT, NTHREADS, 262144, stream>>>(
      h0, xseq, WgT, bg, W1b, b1, W2b, b2, W3b, b3, (float*)d_out);
}
